// Experts_62388694942285
// MI455X (gfx1250) — compile-verified, matrix operand order checked by reading, not run on hardware
//
#include <hip/hip_runtime.h>
#include <hip/hip_bf16.h>

// ---------------- problem constants ----------------
#define T_  1024
#define D_  2048
#define F_  8192
#define E_  8
#define R_  16
#define SCALING_ 2.0f

#define KC_        64   // K elements staged per pipeline stage
#define LSTRIDE_   72   // LDS row stride in halfs (64 + 8 pad -> 144B, 16B aligned)

typedef _Float16 v16h __attribute__((ext_vector_type(16)));
typedef _Float16 v8h  __attribute__((ext_vector_type(8)));
typedef float    v8f  __attribute__((ext_vector_type(8)));
typedef unsigned int u32x4 __attribute__((ext_vector_type(4)));
typedef int          i32x4 __attribute__((ext_vector_type(4)));
typedef int          i32x8 __attribute__((ext_vector_type(8)));

#if defined(__has_builtin)
#  if __has_builtin(__builtin_amdgcn_tensor_load_to_lds) && \
      __has_builtin(__builtin_amdgcn_s_wait_tensorcnt)
#    define USE_TDM 1
#  else
#    define USE_TDM 0
#  endif
#else
#  define USE_TDM 0
#endif

// ---------------- helpers ----------------
__device__ __forceinline__ float gelu_new_f(float u) {
  float u3 = u * u * u;
  float t  = tanhf(0.7978845608028654f * (u + 0.044715f * u3));
  return 0.5f * u * (1.0f + t);
}

// Load one 16x32 (16-bit) WMMA A/B^T fragment from a padded LDS tile.
// Per ISA: lane m (0-15) holds K=off..off+7 and off+16..off+23 of row m;
// lane 16+m holds K=off+8..off+15 and off+24..off+31.
__device__ __forceinline__ v16h load_frag64(const _Float16* base, int row, int off) {
  const v8h* p = (const v8h*)(base + row * LSTRIDE_ + off);
  v8h lo = p[0];
  v8h hi = p[2];
  v16h f;
#pragma unroll
  for (int i = 0; i < 8; ++i) { f[i] = lo[i]; f[i + 8] = hi[i]; }
  return f;
}

#if USE_TDM
// Issue a TDM 2-D tile load: tile_x (contiguous halfs) x tile_y rows, row
// stride `stride_elems` halfs in global memory, into LDS with 4-DWORD padding
// inserted every 32 DWORDs (i.e. LDS row stride = 72 halfs = LSTRIDE_).
__device__ __forceinline__ void tdm_load_tile(const _Float16* gptr, unsigned lds_off,
                                              unsigned tile_x, unsigned tile_y,
                                              unsigned stride_elems) {
  unsigned long long ga = (unsigned long long)(uintptr_t)gptr;
  u32x4 g0;
  g0[0] = 1u;                                            // count=1, user D#
  g0[1] = lds_off;                                       // LDS byte address
  g0[2] = (unsigned)(ga & 0xffffffffu);                  // global addr [31:0]
  g0[3] = (unsigned)((ga >> 32) & 0x01ffffffu) | (2u << 30);  // addr[56:32] | type=2
  i32x8 g1;
  // workgroup_mask=0 | data_size=1(2B)<<16 | pad_enable<<20 |
  // pad_interval=4 (32 DWORDs)<<22 | pad_amount=3 (4 DWORDs)<<25
  g1[0] = (int)((1u << 16) | (1u << 20) | (4u << 22) | (3u << 25));
  g1[1] = (int)((tile_x & 0xffffu) << 16);                       // tensor_dim0 lo16
  g1[2] = (int)(((tile_x >> 16) & 0xffffu) | ((tile_y & 0xffffu) << 16)); // d0 hi | d1 lo
  g1[3] = (int)(((tile_y >> 16) & 0xffffu) | ((tile_x & 0xffffu) << 16)); // d1 hi | tile_dim0
  g1[4] = (int)(tile_y & 0xffffu);                               // tile_dim1, tile_dim2=0
  g1[5] = (int)stride_elems;                                     // tensor_dim0_stride lo32
  g1[6] = 0;                                                     // stride hi | dim1_stride lo
  g1[7] = 0;
  i32x4 z4 = {0, 0, 0, 0};
#if defined(__clang_major__) && (__clang_major__ >= 23)
  i32x8 z8 = {0, 0, 0, 0, 0, 0, 0, 0};
  __builtin_amdgcn_tensor_load_to_lds(g0, g1, z4, z4, z8, 0);
#else
  __builtin_amdgcn_tensor_load_to_lds(g0, g1, z4, z4, 0);
#endif
}
#endif  // USE_TDM

// Synchronous fallback staging: 256 threads, A = 32x64 halfs (1 b128 each),
// B = 64x64 halfs (2 b128 each). Fully uniform, no divergence.
__device__ __forceinline__ void stage_issue_sync(const _Float16* __restrict__ gA,
                                                 const _Float16* __restrict__ gB,
                                                 _Float16* lA, _Float16* lB,
                                                 int tid, int stride_elems) {
  {
    int row = tid >> 3, q = tid & 7;
    *(v8h*)&lA[row * LSTRIDE_ + q * 8] = *(const v8h*)&gA[(size_t)row * stride_elems + q * 8];
  }
#pragma unroll
  for (int s = 0; s < 2; ++s) {
    int slot = tid + s * 256;
    int row = slot >> 3, q = slot & 7;
    *(v8h*)&lB[row * LSTRIDE_ + q * 8] = *(const v8h*)&gB[(size_t)row * stride_elems + q * 8];
  }
}

// ---------------- kernel 1: fp32 -> fp16 convert ----------------
__global__ __launch_bounds__(256) void cvt_f32_f16(const float* __restrict__ in,
                                                   _Float16* __restrict__ out, int n) {
  for (int i = blockIdx.x * 256 + threadIdx.x; i < n; i += gridDim.x * 256)
    out[i] = (_Float16)in[i];
}

// ---------------- kernel 2: transpose + convert (in[Rr,Cc] -> out[Cc,Rr] f16) -----
__global__ __launch_bounds__(256) void transpose_cvt(const float* __restrict__ in,
                                                     _Float16* __restrict__ out,
                                                     int Rr, int Cc) {
  __shared__ float tile[32][33];
  int bx = blockIdx.x * 32;
  int by = blockIdx.y * 32;
  int tx = threadIdx.x, ty = threadIdx.y;  // 32 x 8
  for (int i = ty; i < 32; i += 8) {
    int r = by + i, c = bx + tx;
    tile[i][tx] = (r < Rr && c < Cc) ? in[(size_t)r * Cc + c] : 0.f;
  }
  __syncthreads();
  for (int i = ty; i < 32; i += 8) {
    int r = bx + i, c = by + tx;
    if (r < Cc && c < Rr) out[(size_t)r * Rr + c] = (_Float16)tile[tx][i];
  }
}

// ---------------- kernel 3: router (one wave per token) ----------------
__global__ __launch_bounds__(32) void router_kernel(const float* __restrict__ x,
                                                    const float* __restrict__ Wr,
                                                    const float* __restrict__ br,
                                                    float* __restrict__ combine,
                                                    float* __restrict__ wsum) {
  int t = blockIdx.x;
  int lane = threadIdx.x;
  float acc[E_];
#pragma unroll
  for (int e = 0; e < E_; ++e) acc[e] = 0.f;
  const float* xr = x + (size_t)t * D_;
  for (int d = lane; d < D_; d += 32) {
    float xv = xr[d];
    const float* wr = Wr + (size_t)d * E_;
#pragma unroll
    for (int e = 0; e < E_; ++e) acc[e] += xv * wr[e];
  }
#pragma unroll
  for (int off = 16; off >= 1; off >>= 1) {
#pragma unroll
    for (int e = 0; e < E_; ++e) acc[e] += __shfl_xor(acc[e], off, 32);
  }
  if (lane == 0) {
    float l[E_], p[E_];
    float mx = -1e30f;
#pragma unroll
    for (int e = 0; e < E_; ++e) { l[e] = acc[e] + br[e]; mx = fmaxf(mx, l[e]); }
    float s = 0.f;
#pragma unroll
    for (int e = 0; e < E_; ++e) { p[e] = expf(l[e] - mx); s += p[e]; }
    float inv = 1.0f / s;
#pragma unroll
    for (int e = 0; e < E_; ++e) p[e] *= inv;
    int i1 = 0;
#pragma unroll
    for (int e = 1; e < E_; ++e) if (p[e] > p[i1]) i1 = e;
    int i2 = (i1 == 0) ? 1 : 0;
#pragma unroll
    for (int e = 0; e < E_; ++e) if (e != i1 && p[e] > p[i2]) i2 = e;
#pragma unroll
    for (int e = 0; e < E_; ++e)
      combine[t * E_ + e] = (e == i1) ? p[i1] : ((e == i2) ? p[i2] : 0.f);
    wsum[t] = p[i1] + p[i2];
  }
}

// ---------------- kernel 4: t1[t, e*16+r] = sum_d x[t,d] * A1[e,r,d]  (fp32) ------
__global__ __launch_bounds__(256) void t1_kernel(const float* __restrict__ x,
                                                 const float* __restrict__ A1,
                                                 float* __restrict__ t1) {
  __shared__ float Xs[16][64];
  __shared__ float As[128][65];
  int tid = threadIdx.x;
  int t0 = blockIdx.x * 16;
  int tl = tid & 15, g = tid >> 4;
  float acc[8];
#pragma unroll
  for (int j = 0; j < 8; ++j) acc[j] = 0.f;
  for (int d0 = 0; d0 < D_; d0 += 64) {
    for (int idx = tid; idx < 16 * 64; idx += 256) {
      int rr = idx >> 6, cc = idx & 63;
      Xs[rr][cc] = x[(size_t)(t0 + rr) * D_ + d0 + cc];
    }
    for (int idx = tid; idx < 128 * 64; idx += 256) {
      int rr = idx >> 6, cc = idx & 63;
      As[rr][cc] = A1[(size_t)rr * D_ + d0 + cc];
    }
    __syncthreads();
    for (int dd = 0; dd < 64; ++dd) {
      float xv = Xs[tl][dd];
#pragma unroll
      for (int j = 0; j < 8; ++j) acc[j] += xv * As[g * 8 + j][dd];
    }
    __syncthreads();
  }
#pragma unroll
  for (int j = 0; j < 8; ++j) t1[(size_t)(t0 + tl) * 128 + g * 8 + j] = acc[j];
}

// ---------------- kernel 5: fused FC1 + LoRA1 + gelu + combine + t2-partials ------
// grid: (F_/1024 = 8, T_/32 = 32), block 256 (8 waves).
__global__ __launch_bounds__(256) void fc1_fused(
    const _Float16* __restrict__ xh, const _Float16* __restrict__ w1t,
    const float* __restrict__ b1, const float* __restrict__ t1,
    const float* __restrict__ B1, const float* __restrict__ A2,
    const float* __restrict__ combine, _Float16* __restrict__ mixh,
    float* __restrict__ t2part) {
  __shared__ _Float16 As[2][32 * LSTRIDE_];
  __shared__ _Float16 Bs[2][64 * LSTRIDE_];
  __shared__ float t1s[32 * 128];
  __shared__ float cmbs[32 * 8];
  __shared__ float b1s[64];
  __shared__ float B1s[64 * 128];     // [f_local][e*16+r]
  __shared__ float cas[32 * 8 * 64];  // [t_local][e][f_local]

  const int tid = threadIdx.x;
  const int t0  = blockIdx.y * 32;
  const int fg  = blockIdx.x;
  const int fg0 = fg * 1024;

  for (int idx = tid; idx < 32 * 128; idx += 256) t1s[idx] = t1[(size_t)t0 * 128 + idx];
  for (int idx = tid; idx < 32 * 8; idx += 256)   cmbs[idx] = combine[(size_t)t0 * E_ + idx];

  const int w = tid >> 5, lane = tid & 31;
  const int m_base = (w & 1) * 16;
  const int f_sub  = (w >> 1) * 16;
  const int lm  = lane & 15;
  const int ko  = (lane & 16) ? 8 : 0;
  const int sel = lane >> 4;
  const int f_local = f_sub + lm;

  const int tl2 = tid >> 3, e2 = tid & 7;
  float t2acc[R_];
#pragma unroll
  for (int r = 0; r < R_; ++r) t2acc[r] = 0.f;

  const _Float16* gA = xh + (size_t)t0 * D_;

  __syncthreads();

  for (int ci = 0; ci < 16; ++ci) {
    const int fb = fg0 + ci * 64;
    if (tid < 64) b1s[tid] = b1[fb + tid];
    for (int idx = tid; idx < 64 * 128; idx += 256) {
      int fl = idx >> 7, q = idx & 127;
      int e = q >> 4, r = q & 15;
      B1s[fl * 128 + q] = B1[((size_t)e * F_ + (fb + fl)) * R_ + r];
    }
    const _Float16* gB = w1t + (size_t)fb * D_;

    // pipeline prologue: stage 0 into buffer 0
#if USE_TDM
    if (tid < 32) {
      tdm_load_tile(gA, (unsigned)(uintptr_t)&As[0][0], KC_, 32, D_);
      tdm_load_tile(gB, (unsigned)(uintptr_t)&Bs[0][0], KC_, 64, D_);
    }
#else
    stage_issue_sync(gA, gB, &As[0][0], &Bs[0][0], tid, D_);
#endif

    v8f acc = {};
    for (int k0 = 0; k0 < D_; k0 += KC_) {
      const int cur = (k0 >> 6) & 1;
      const int nxt = cur ^ 1;
      const bool more = (k0 + KC_ < D_);
#if USE_TDM
      if (tid < 32) {
        if (more) {
          tdm_load_tile(gA + (k0 + KC_), (unsigned)(uintptr_t)&As[nxt][0], KC_, 32, D_);
          tdm_load_tile(gB + (k0 + KC_), (unsigned)(uintptr_t)&Bs[nxt][0], KC_, 64, D_);
          __builtin_amdgcn_s_wait_tensorcnt((short)2);
        } else {
          __builtin_amdgcn_s_wait_tensorcnt((short)0);
        }
      }
#else
      if (more)
        stage_issue_sync(gA + (k0 + KC_), gB + (k0 + KC_), &As[nxt][0], &Bs[nxt][0], tid, D_);
#endif
      __syncthreads();
#pragma unroll
      for (int s = 0; s < 2; ++s) {
        v16h fa  = load_frag64(&As[cur][0], m_base + lm, s * 32 + ko);
        v16h fbv = load_frag64(&Bs[cur][0], f_sub + lm, s * 32 + ko);
        acc = __builtin_amdgcn_wmma_f32_16x16x32_f16(false, fa, false, fbv,
                                                     (short)0, acc, false, false);
      }
      __syncthreads();
    }

    // fp32 epilogue: base + LoRA1, gelu, combine-weight, stage ca, write mixh
#pragma unroll
    for (int j = 0; j < 8; ++j) {
      int tl = m_base + sel * 8 + j;
      float v = acc[j] + b1s[f_local];
      const float* t1p = &t1s[tl * 128];
      const float* b1p = &B1s[f_local * 128];
      float mixv = 0.f;
#pragma unroll
      for (int e = 0; e < E_; ++e) {
        float dot = 0.f;
#pragma unroll
        for (int r = 0; r < R_; ++r) dot += t1p[e * 16 + r] * b1p[e * 16 + r];
        float u   = v + SCALING_ * dot;
        float g   = gelu_new_f(u);
        float cae = cmbs[tl * 8 + e] * g;
        cas[tl * 512 + e * 64 + f_local] = cae;
        mixv += cae;
      }
      mixh[(size_t)(t0 + tl) * F_ + fb + f_local] = (_Float16)mixv;
    }
    __syncthreads();

    // t2 partial reduction over this 64-wide f chunk
    {
      const float* cp = &cas[tl2 * 512 + e2 * 64];
#pragma unroll
      for (int r = 0; r < R_; ++r) {
        const float* a2p = &A2[((size_t)e2 * R_ + r) * F_ + fb];
        float s = 0.f;
        for (int f = 0; f < 64; ++f) s += cp[f] * a2p[f];
        t2acc[r] += s;
      }
    }
    __syncthreads();
  }

#pragma unroll
  for (int r = 0; r < R_; ++r)
    t2part[(size_t)fg * T_ * 128 + (size_t)(t0 + tl2) * 128 + e2 * 16 + r] = t2acc[r];
}

// ---------------- kernel 6: out = mix@W2 + wsum*b2 + SCALING*(t2 . B2) ------------
// grid: (D_/64 = 32, T_/32 = 32), block 256 (8 waves).
__global__ __launch_bounds__(256) void out_gemm(
    const _Float16* __restrict__ mixh, const _Float16* __restrict__ w2t,
    const float* __restrict__ b2, const float* __restrict__ t2part,
    const float* __restrict__ B2, const float* __restrict__ wsum,
    float* __restrict__ out) {
  __shared__ _Float16 As[2][32 * LSTRIDE_];
  __shared__ _Float16 Bs[2][64 * LSTRIDE_];
  __shared__ float t2s[32 * 128];
  __shared__ float B2s[64 * 128];
  __shared__ float b2s[64];
  __shared__ float wss[32];

  const int tid = threadIdx.x;
  const int t0 = blockIdx.y * 32;
  const int d0 = blockIdx.x * 64;

  for (int idx = tid; idx < 32 * 128; idx += 256) {
    float s = 0.f;
#pragma unroll
    for (int g = 0; g < 8; ++g)
      s += t2part[(size_t)g * T_ * 128 + (size_t)t0 * 128 + idx];
    t2s[idx] = s;
  }
  for (int idx = tid; idx < 64 * 128; idx += 256) {
    int dl = idx >> 7, q = idx & 127;
    int e = q >> 4, r = q & 15;
    B2s[dl * 128 + q] = B2[((size_t)e * D_ + (d0 + dl)) * R_ + r];
  }
  if (tid < 64) b2s[tid] = b2[d0 + tid];
  if (tid < 32) wss[tid] = wsum[t0 + tid];

  const int w = tid >> 5, lane = tid & 31;
  const int m_base = (w & 1) * 16, d_sub = (w >> 1) * 16;
  const int lm = lane & 15, ko = (lane & 16) ? 8 : 0;
  const int sel = lane >> 4;
  const int d_local = d_sub + lm;

  const _Float16* gA = mixh + (size_t)t0 * F_;
  const _Float16* gB = w2t + (size_t)d0 * F_;

#if USE_TDM
  if (tid < 32) {
    tdm_load_tile(gA, (unsigned)(uintptr_t)&As[0][0], KC_, 32, F_);
    tdm_load_tile(gB, (unsigned)(uintptr_t)&Bs[0][0], KC_, 64, F_);
  }
#else
  stage_issue_sync(gA, gB, &As[0][0], &Bs[0][0], tid, F_);
#endif

  v8f acc = {};
  __syncthreads();
  for (int k0 = 0; k0 < F_; k0 += KC_) {
    const int cur = (k0 >> 6) & 1;
    const int nxt = cur ^ 1;
    const bool more = (k0 + KC_ < F_);
#if USE_TDM
    if (tid < 32) {
      if (more) {
        tdm_load_tile(gA + (k0 + KC_), (unsigned)(uintptr_t)&As[nxt][0], KC_, 32, F_);
        tdm_load_tile(gB + (k0 + KC_), (unsigned)(uintptr_t)&Bs[nxt][0], KC_, 64, F_);
        __builtin_amdgcn_s_wait_tensorcnt((short)2);
      } else {
        __builtin_amdgcn_s_wait_tensorcnt((short)0);
      }
    }
#else
    if (more)
      stage_issue_sync(gA + (k0 + KC_), gB + (k0 + KC_), &As[nxt][0], &Bs[nxt][0], tid, F_);
#endif
    __syncthreads();
#pragma unroll
    for (int s = 0; s < 2; ++s) {
      v16h fa = load_frag64(&As[cur][0], m_base + lm, s * 32 + ko);
      v16h fb = load_frag64(&Bs[cur][0], d_sub + lm, s * 32 + ko);
      acc = __builtin_amdgcn_wmma_f32_16x16x32_f16(false, fa, false, fb,
                                                   (short)0, acc, false, false);
    }
    __syncthreads();
  }

#pragma unroll
  for (int j = 0; j < 8; ++j) {
    int tl = m_base + sel * 8 + j;
    float v = acc[j] + wss[tl] * b2s[d_local];
    const float* tp = &t2s[tl * 128];
    const float* bp = &B2s[d_local * 128];
    float dot = 0.f;
#pragma unroll
    for (int q = 0; q < 128; ++q) dot += tp[q] * bp[q];
    v += SCALING_ * dot;
    out[(size_t)(t0 + tl) * D_ + d0 + d_local] = v;
  }
}

// ---------------- launch ----------------
extern "C" void kernel_launch(void* const* d_in, const int* in_sizes, int n_in,
                              void* d_out, int out_size, void* d_ws, size_t ws_size,
                              hipStream_t stream) {
  (void)in_sizes; (void)n_in; (void)out_size; (void)ws_size;
  const float* x  = (const float*)d_in[0];
  const float* Wr = (const float*)d_in[1];
  const float* br = (const float*)d_in[2];
  const float* W1 = (const float*)d_in[3];
  const float* b1 = (const float*)d_in[4];
  const float* W2 = (const float*)d_in[5];
  const float* b2 = (const float*)d_in[6];
  const float* A1 = (const float*)d_in[7];
  const float* B1 = (const float*)d_in[8];
  const float* A2 = (const float*)d_in[9];
  const float* B2 = (const float*)d_in[10];
  float* out = (float*)d_out;

  char* ws = (char*)d_ws;
  size_t off = 0;
  auto alloc = [&](size_t bytes) -> char* {
    char* p = ws + off;
    off += (bytes + 255) & ~(size_t)255;
    return p;
  };
  _Float16* xh   = (_Float16*)alloc((size_t)T_ * D_ * 2);
  _Float16* w1t  = (_Float16*)alloc((size_t)F_ * D_ * 2);
  _Float16* w2t  = (_Float16*)alloc((size_t)D_ * F_ * 2);
  _Float16* mixh = (_Float16*)alloc((size_t)T_ * F_ * 2);
  float* combine = (float*)alloc((size_t)T_ * E_ * 4);
  float* wsum    = (float*)alloc((size_t)T_ * 4);
  float* t1buf   = (float*)alloc((size_t)T_ * E_ * R_ * 4);
  float* t2part  = (float*)alloc((size_t)8 * T_ * E_ * R_ * 4);

  cvt_f32_f16<<<512, 256, 0, stream>>>(x, xh, T_ * D_);
  transpose_cvt<<<dim3(F_ / 32, D_ / 32), dim3(32, 8), 0, stream>>>(W1, w1t, D_, F_);
  transpose_cvt<<<dim3(D_ / 32, F_ / 32), dim3(32, 8), 0, stream>>>(W2, w2t, F_, D_);
  router_kernel<<<T_, 32, 0, stream>>>(x, Wr, br, combine, wsum);
  t1_kernel<<<T_ / 16, 256, 0, stream>>>(x, A1, t1buf);
  fc1_fused<<<dim3(F_ / 1024, T_ / 32), 256, 0, stream>>>(
      xh, w1t, b1, t1buf, B1, A2, combine, mixh, t2part);
  out_gemm<<<dim3(D_ / 64, T_ / 32), 256, 0, stream>>>(
      mixh, w2t, b2, t2part, B2, wsum, out);
}
